// MultiAssetLSTM_38594576122293
// MI455X (gfx1250) — matrix operand order repaired and checked by reading, not compiled, not run on hardware
//
#include <hip/hip_runtime.h>

// MultiAssetLSTM for MI455X (gfx1250, wave32).
// B=512, T=512, I=32, H=64, NUM_TICKERS=8.
//
// Latency-bound sequential recurrence. One workgroup = 16 batch rows,
// 4 waves. Wave w owns h-units [16w,16w+16) and keeps the i/f/g/o gate
// tiles as f32 16x16 WMMA accumulators (gate math fully in registers).
// Weight B-fragments live in registers for the whole T loop.
//
// Software pipeline (1 barrier/step):
//   step t:  h-chain  = wmma(hA0)->wmma(hA1) seeded with accx(t)   [critical]
//            accx(t+1)= x_{t+1} @ w_ih^T (inline-0 C; xA from LDS) [off-path]
//            stage x(t+2) -> LDS, preload x(t+3) from global       [off-path]
// Bias folded into the gate nonlinearity argument (free for sigmoid gates
// via v_fma). Gate nonlinearities use the CDNA5 V_TANH_F32 trans op.
// T loop unrolled by 2 so double-buffer parities are compile-time constant.

typedef _Float16 v16h __attribute__((ext_vector_type(16)));
typedef float    v8f  __attribute__((ext_vector_type(8)));

#define TLEN 512
#define BSZ  512
#define ISZ  32
#define HSZ  64
#define NTK  8
#define MTILE 16   // batch rows per workgroup

// CDNA5 hardware tanh. Prefer the builtin (scheduler handles the TRANS
// hazard with s_delay_alu and can dual-issue); fall back to inline asm with
// an explicit v_nop for the 1-instruction TRANS gap (ISA §7.4).
static __device__ __forceinline__ float fast_tanh(float x) {
#if __has_builtin(__builtin_amdgcn_tanhf)
    return __builtin_amdgcn_tanhf(x);
#else
    float y;
    asm("v_tanh_f32 %0, %1\n\tv_nop" : "=v"(y) : "v"(x));
    return y;
#endif
}
// sigmoid(x + b) = 0.5 + 0.5*tanh(0.5*x + 0.5*b): arg is a single v_fma.
static __device__ __forceinline__ float sigmoid_biased(float x, float halfbias) {
    return 0.5f + 0.5f * fast_tanh(__builtin_fmaf(0.5f, x, halfbias));
}

__global__ __launch_bounds__(128, 1)
void lstm_wmma_kernel(const float* __restrict__ x,
                      const float* __restrict__ w_ih,
                      const float* __restrict__ w_hh,
                      const float* __restrict__ b_ih,
                      const float* __restrict__ b_hh,
                      const float* __restrict__ w_fc,
                      const float* __restrict__ b_fc,
                      float* __restrict__ out)
{
    __shared__ _Float16 xs[2][MTILE * ISZ];    // double-buffered x tile (f16)
    __shared__ _Float16 hb[2][MTILE * HSZ];    // double-buffered h state (f16)

    const int tid   = threadIdx.x;
    const int wave  = tid >> 5;     // 0..3
    const int lane  = tid & 31;     // wave32
    const int mbase = blockIdx.x * MTILE;
    const int j0    = wave * 16;    // this wave's h-unit tile base
    const int nrow  = lane & 15;    // N index within tile (B/C/D)
    const int khalf = lane >> 4;    // K-half select in fragment layouts

    // ---------------- one-time: gather weight B-fragments into registers ---
    // B is KxN; element (k,n) = w[n][k]. Lane holds column n = lane&15;
    // half d holds k = 16*(lane>>4) + d.
    v16h Bx[4];        // w_ih^T tiles (K=32), one per gate
    v16h Bh[4][2];     // w_hh^T tiles (K=64 -> two K=32 fragments), per gate
    float hb_i, hb_f, hb_o, b_g;   // 0.5*bias for sigmoid gates, bias for g
    {
        float bias[4];
        #pragma unroll
        for (int g = 0; g < 4; ++g) {
            const int n = g * HSZ + j0 + nrow;      // column in 4H space
            bias[g] = b_ih[n] + b_hh[n];
            #pragma unroll
            for (int d = 0; d < 16; ++d)
                Bx[g][d] = (_Float16)w_ih[n * ISZ + (khalf << 4) + d];
            #pragma unroll
            for (int f = 0; f < 2; ++f)
                #pragma unroll
                for (int d = 0; d < 16; ++d)
                    Bh[g][f][d] = (_Float16)w_hh[n * HSZ + (f << 5) + (khalf << 4) + d];
        }
        hb_i = 0.5f * bias[0];
        hb_f = 0.5f * bias[1];
        b_g  = bias[2];
        hb_o = 0.5f * bias[3];
    }

    // A-fragment gather pattern (16-bit A 16x32, contiguous k order chosen to match B):
// half d -> k = 16*(lane>=16) + d, row m = lane&15.
    int akofs[16];
    #pragma unroll
    for (int d = 0; d < 16; ++d)
        akofs[d] = (khalf << 4) + d;
    const int am = lane & 15;

    // x staging: thread's fixed (row, col) slot; 128 threads * 4 floats = 16x32
    const int srow = tid >> 3;
    const int scol = (tid & 7) << 2;
    const float* xp = &x[(size_t)(mbase + srow) * TLEN * ISZ + scol];

    // zero both h buffers and the c state
    for (int i = tid; i < 2 * MTILE * HSZ; i += 128) hb[0][i] = (_Float16)0.0f;
    float c[8];
    #pragma unroll
    for (int r = 0; r < 8; ++r) c[r] = 0.0f;

    // prologue: stage x(0), x(1); compute accx(0); preload x(2)
    {
        const float4 a0 = *(const float4*)(xp);
        const float4 a1 = *(const float4*)(xp + ISZ);
        _Float16* p0 = &xs[0][srow * ISZ + scol];
        _Float16* p1 = &xs[1][srow * ISZ + scol];
        p0[0]=(_Float16)a0.x; p0[1]=(_Float16)a0.y; p0[2]=(_Float16)a0.z; p0[3]=(_Float16)a0.w;
        p1[0]=(_Float16)a1.x; p1[1]=(_Float16)a1.y; p1[2]=(_Float16)a1.z; p1[3]=(_Float16)a1.w;
    }
    __syncthreads();

    const v8f zeroC = {};          // lowered to inline-constant 0 C operand
    v8f accx[4];                   // x_t @ w_ih^T, computed one step ahead
    {
        v16h xA;
        #pragma unroll
        for (int d = 0; d < 16; ++d) xA[d] = xs[0][am * ISZ + akofs[d]];
        #pragma unroll
        for (int g = 0; g < 4; ++g)
            accx[g] = __builtin_amdgcn_wmma_f32_16x16x32_f16(
                          false, xA, false, Bx[g], (short)0, zeroC, false, false);
    }
    float4 v = *(const float4*)(xp + 2 * ISZ);   // x(2)

    // ---------------- sequential recurrence over T --------------------------
    // Unroll by 2: buffer parities become compile-time constants, so all LDS
    // base addresses are loop-invariant (no per-step parity address math).
    #pragma unroll 2
    for (int t = 0; t < TLEN; ++t) {
        _Float16* hR  = hb[(t + 1) & 1];        // h_{t-1} (written at t-1)
        _Float16* hW  = hb[t & 1];              // h_t target
        _Float16* xsN = xs[(t + 1) & 1];        // x_{t+1} (staged at t-1)
        _Float16* xsW = xs[t & 1];              // x_{t+2} staging target

        // ---- critical chain: h gather -> 2 WMMAs per gate ----
        v16h hA0, hA1;
        #pragma unroll
        for (int d = 0; d < 16; ++d) hA0[d] = hR[am * HSZ + akofs[d]];
        #pragma unroll
        for (int d = 0; d < 16; ++d) hA1[d] = hR[am * HSZ + 32 + akofs[d]];

        v8f acc[4];
        #pragma unroll
        for (int g = 0; g < 4; ++g) {
            acc[g] = __builtin_amdgcn_wmma_f32_16x16x32_f16(
                         false, hA0, false, Bh[g][0], (short)0, accx[g], false, false);
            acc[g] = __builtin_amdgcn_wmma_f32_16x16x32_f16(
                         false, hA1, false, Bh[g][1], (short)0, acc[g], false, false);
        }

        // ---- off-path: accx(t+1) = x_{t+1} @ w_ih^T (inline-0 C) ----
        {
            v16h xA;
            #pragma unroll
            for (int d = 0; d < 16; ++d) xA[d] = xsN[am * ISZ + akofs[d]];
            #pragma unroll
            for (int g = 0; g < 4; ++g)
                accx[g] = __builtin_amdgcn_wmma_f32_16x16x32_f16(
                              false, xA, false, Bx[g], (short)0, zeroC, false, false);
        }

        // ---- off-path: stage x(t+2) (preloaded), issue load of x(t+3) ----
        {
            _Float16* p = &xsW[srow * ISZ + scol];
            p[0]=(_Float16)v.x; p[1]=(_Float16)v.y; p[2]=(_Float16)v.z; p[3]=(_Float16)v.w;
        }
        const int tn = (t + 3 < TLEN) ? (t + 3) : (TLEN - 1);
        v = *(const float4*)(xp + (size_t)tn * ISZ);

        // ---- gate math (V_TANH_F32), bias folded in, c update, h store ----
        // D layout: VGPR r -> M = r + 8*(lane>=16), N = lane&15.
        #pragma unroll
        for (int r = 0; r < 8; ++r) {
            const float si = sigmoid_biased(acc[0][r], hb_i);
            const float sf = sigmoid_biased(acc[1][r], hb_f);
            const float tg = fast_tanh(acc[2][r] + b_g);
            const float so = sigmoid_biased(acc[3][r], hb_o);
            const float cn = sf * c[r] + si * tg;
            c[r] = cn;
            const float h = so * fast_tanh(cn);
            const int m = r + (khalf << 3);
            hW[m * HSZ + j0 + nrow] = (_Float16)h;
        }
        __syncthreads();   // h_t and x_{t+2} visible for next step
    }

    // ---------------- fused FC: out[b,q] = h_last[b,:] . w_fc[q,:] + b_fc[q]
    {
        const _Float16* hLast = hb[(TLEN - 1) & 1];
        const int b = tid >> 3;   // 0..15
        const int q = tid & 7;    // 0..7
        float s = b_fc[q];
        #pragma unroll
        for (int j = 0; j < HSZ; ++j)
            s += (float)hLast[b * HSZ + j] * w_fc[q * HSZ + j];
        out[(size_t)(mbase + b) * NTK + q] = s;
    }
}

extern "C" void kernel_launch(void* const* d_in, const int* in_sizes, int n_in,
                              void* d_out, int out_size, void* d_ws, size_t ws_size,
                              hipStream_t stream) {
    (void)in_sizes; (void)n_in; (void)d_ws; (void)ws_size; (void)out_size;
    const float* x    = (const float*)d_in[0];
    const float* w_ih = (const float*)d_in[1];
    const float* w_hh = (const float*)d_in[2];
    const float* b_ih = (const float*)d_in[3];
    const float* b_hh = (const float*)d_in[4];
    const float* w_fc = (const float*)d_in[5];
    const float* b_fc = (const float*)d_in[6];
    float* out = (float*)d_out;

    dim3 grid(BSZ / MTILE);   // 32 workgroups
    dim3 block(128);          // 4 waves (wave32)
    lstm_wmma_kernel<<<grid, block, 0, stream>>>(x, w_ih, w_hh, b_ih, b_hh,
                                                 w_fc, b_fc, out);
}
